// HGT_47794396070375
// MI455X (gfx1250) — matrix operand order repaired and checked by reading, not compiled, not run on hardware
//
#include <hip/hip_runtime.h>
#include <hip/hip_bf16.h>
#include <math.h>

// ---------------------------------------------------------------------------
// HGT forward for MI455X (gfx1250, wave32, WMMA).
// Dense 128-wide GEMMs -> v_wmma_f32_16x16x32_bf16 (bf16 in, f32 acc).
// Edge attention / segment softmax / scatter -> L2-resident VALU kernels.
// Relation-specific per-head 32x32 transforms folded into composite 128x128
// weights so kr/vr are single WMMA GEMMs from h directly.
// ---------------------------------------------------------------------------

typedef __attribute__((ext_vector_type(16))) __bf16 v16bf;
typedef __attribute__((ext_vector_type(8)))  float  v8f;

#define HGT_H 4
#define HGT_D 32
#define HGT_HID 128

__device__ __forceinline__ unsigned short f2bf(float f) {
    unsigned int u = __float_as_uint(f);
    unsigned int r = 0x7fffu + ((u >> 16) & 1u);   // round-to-nearest-even
    u += r;
    return (unsigned short)(u >> 16);
}

__device__ __forceinline__ unsigned int f2key(float f) {
    unsigned int b = __float_as_uint(f);
    return (b & 0x80000000u) ? ~b : (b | 0x80000000u);
}
__device__ __forceinline__ float key2f(unsigned int k) {
    unsigned int b = (k & 0x80000000u) ? (k ^ 0x80000000u) : ~k;
    return __uint_as_float(b);
}

__device__ __forceinline__ float gelu_exact(float x) {
    return 0.5f * x * (1.0f + erff(x * 0.70710678118654752440f));
}

// ---------------------------------------------------------------------------
// Generic 128-wide GEMM:  Y = epilogue( act_in(X) @ W + bias )
//   X:    [nrows,128] f32, row stride ldx
//   W:    [128,128]   f32 row-major (k-major), converted to bf16^T in LDS
//   epilogue: optional ReLU, optional sigmoid(skip)-gated residual blend,
//             dual stores (Y1 contiguous, Y2 strided slice of d_out).
// Block = 256 threads = 8 waves; 32 rows x 128 cols per block.
// Each wave owns one 16-row band and two 16-col tiles; K-loop of 4 x 32.
// ---------------------------------------------------------------------------
__global__ void __launch_bounds__(256)
gemm128_kernel(const float* __restrict__ X, long long ldx,
               const float* __restrict__ W,
               const float* __restrict__ bias,
               float* __restrict__ Y1, long long ldy1,
               float* __restrict__ Y2, long long ldy2,
               const float* __restrict__ resid, long long ldr,
               const float* __restrict__ skipv,
               int nrows, int act_in, int act_out)
{
    __shared__ unsigned short lsA[32][136];    // bf16 bits, activations
    __shared__ unsigned short lsB[128][136];   // bf16 bits, W transposed [n][k]

    const int tid = threadIdx.x;
    const long long row0 = (long long)blockIdx.x * 32;

    // Stage W^T (bf16) into LDS
    for (int idx = tid; idx < 128 * 128; idx += 256) {
        int k = idx >> 7, n = idx & 127;
        lsB[n][k] = f2bf(W[idx]);
    }
    // Stage activation tile (f32 -> optional GELU -> bf16)
    for (int idx = tid; idx < 32 * 128; idx += 256) {
        int r = idx >> 7, k = idx & 127;
        long long gr = row0 + r;
        float v = (gr < (long long)nrows) ? X[gr * ldx + k] : 0.0f;
        if (act_in == 2) v = gelu_exact(v);
        lsA[r][k] = f2bf(v);
    }
    __syncthreads();

    const int wave  = tid >> 5;
    const int lane  = tid & 31;
    const int band  = wave >> 2;        // 0..1 : rows band*16 .. +16
    const int cpair = wave & 3;         // cols cpair*32 .. +32 (2 tiles)
    const int half  = lane >> 4;        // lane group
    const int l16   = lane & 15;

    v8f c0 = {}; v8f c1 = {};

    for (int ks = 0; ks < 4; ++ks) {
        // A fragment: 16x32 bf16, M = l16, K-halves selected by lane group
        union { v16bf v; unsigned int u[8]; } A;
        const unsigned short* arow = &lsA[band * 16 + l16][ks * 32];
#pragma unroll
        for (int i = 0; i < 8; ++i) {
            int k0 = ((i & 4) << 2) + (half << 3) + ((i & 3) << 1);
            A.u[i] = *(const unsigned int*)(arow + k0);
        }
        // B fragments: 32x16 bf16, N = l16, same K order as A
        union { v16bf v; unsigned int u[8]; } B0, B1;
        const unsigned short* b0row = &lsB[cpair * 32 + l16][ks * 32 + half * 8];
        const unsigned short* b1row = &lsB[cpair * 32 + 16 + l16][ks * 32 + half * 8];
#pragma unroll
        for (int i = 0; i < 8; ++i) {
            B0.u[i] = *(const unsigned int*)(b0row + ((i & 4) << 2) + ((i & 3) << 1));
            B1.u[i] = *(const unsigned int*)(b1row + ((i & 4) << 2) + ((i & 3) << 1));
        }
        c0 = __builtin_amdgcn_wmma_f32_16x16x32_bf16(false, A.v, false, B0.v,
                                                     (short)0, c0, false, false);
        c1 = __builtin_amdgcn_wmma_f32_16x16x32_bf16(false, A.v, false, B1.v,
                                                     (short)0, c1, false, false);
    }

    const float ablend = skipv ? (1.0f / (1.0f + expf(-*skipv))) : 1.0f;

#pragma unroll
    for (int t = 0; t < 2; ++t) {
        v8f cc = t ? c1 : c0;
        const int ncol = cpair * 32 + t * 16 + l16;
        const float bv = bias ? bias[ncol] : 0.0f;
#pragma unroll
        for (int r = 0; r < 8; ++r) {
            long long grow = row0 + band * 16 + r + half * 8;  // C/D layout
            if (grow < (long long)nrows) {
                float val = cc[r] + bv;
                if (act_out == 1) val = fmaxf(val, 0.0f);
                float res = val;
                if (resid) res = ablend * val + (1.0f - ablend) * resid[grow * ldr + ncol];
                if (Y1) Y1[grow * ldy1 + ncol] = res;
                if (Y2) Y2[grow * ldy2 + ncol] = res;
            }
        }
    }
}

// ---------------------------------------------------------------------------
// Composite weight: Wc[k][n] = sum_d W[k][h*32+d] * rel[h][d][e]
//                   bc[n]    = sum_d b[h*32+d]    * rel[h][d][e]   (h=n/32,e=n%32)
// (== folding the block-diagonal per-head 32x32 relation matrix into W/b)
// ---------------------------------------------------------------------------
__global__ void compose_kernel(const float* __restrict__ W,
                               const float* __restrict__ b,
                               const float* __restrict__ rel,  // [4][32][32]
                               float* __restrict__ Wc,
                               float* __restrict__ bc)
{
    int idx = blockIdx.x * 256 + threadIdx.x;
    if (idx < 16384) {
        int k = idx >> 7, n = idx & 127;
        int h = n >> 5, e = n & 31;
        const float* wrow = W + k * 128 + h * 32;
        const float* rcol = rel + h * 1024 + e;   // stride 32 over d
        float s = 0.0f;
#pragma unroll 8
        for (int d = 0; d < 32; ++d) s += wrow[d] * rcol[d * 32];
        Wc[idx] = s;
    } else if (idx < 16384 + 128) {
        int n = idx - 16384;
        int h = n >> 5, e = n & 31;
        float s = 0.0f;
        for (int d = 0; d < 32; ++d) s += b[h * 32 + d] * rel[h * 1024 + d * 32 + e];
        bc[n] = s;
    }
}

// ---------------------------------------------------------------------------
// Edge kernels (one relation). L2-resident gathers/scatters.
// ---------------------------------------------------------------------------
__global__ void edge_alpha_kernel(const int* __restrict__ src,
                                  const int* __restrict__ dst,
                                  const float* __restrict__ q,    // [Ndt][128]
                                  const float* __restrict__ kr,   // [Nsrc][128]
                                  const float* __restrict__ prel, // [4]
                                  float* __restrict__ alpha,      // [E*4]
                                  unsigned int* __restrict__ amaxKey, // [Ndt*4]
                                  int E)
{
    int idx = blockIdx.x * 256 + threadIdx.x;   // e*4 + h
    if (idx >= E * HGT_H) return;
    int e = idx >> 2, h = idx & 3;
    int s = src[e], d = dst[e];
    const float* qp = q  + (long long)d * 128 + h * 32;
    const float* kp = kr + (long long)s * 128 + h * 32;
    float acc = 0.0f;
#pragma unroll 8
    for (int i = 0; i < 32; ++i) acc += qp[i] * kp[i];
    float a = acc * prel[h] * 0.17677669529663687f;   // 1/sqrt(32)
    alpha[idx] = a;
    atomicMax(&amaxKey[(long long)d * 4 + h], f2key(a));
}

__global__ void edge_exp_kernel(const int* __restrict__ dst,
                                float* __restrict__ alpha,
                                const unsigned int* __restrict__ amaxKey,
                                float* __restrict__ ssum,
                                int E)
{
    int idx = blockIdx.x * 256 + threadIdx.x;
    if (idx >= E * HGT_H) return;
    int e = idx >> 2, h = idx & 3;
    int d = dst[e];
    float m = key2f(amaxKey[(long long)d * 4 + h]);
    if (m != m) m = 0.0f;                    // NaN guard (untouched segment)
    float ex = expf(alpha[idx] - m);
    alpha[idx] = ex;
    atomicAdd(&ssum[(long long)d * 4 + h], ex);
}

// thread per (edge, channel): coalesced vr gather, 1 atomicAdd each
__global__ void edge_scatter_kernel(const int* __restrict__ src,
                                    const int* __restrict__ dst,
                                    const float* __restrict__ vr,
                                    const float* __restrict__ alpha,
                                    const float* __restrict__ ssum,
                                    float* __restrict__ agg,
                                    int E)
{
    long long idx = (long long)blockIdx.x * 256 + threadIdx.x; // e*128 + c
    if (idx >= (long long)E * 128) return;
    int e = (int)(idx >> 7);
    int c = (int)(idx & 127);
    int h = c >> 5;
    int s = src[e], d = dst[e];
    float w = alpha[(long long)e * 4 + h] / (ssum[(long long)d * 4 + h] + 1e-16f);
    atomicAdd(&agg[(long long)d * 128 + c], vr[(long long)s * 128 + c] * w);
}

__global__ void fill_kernel(unsigned int* __restrict__ p, unsigned int v, long long n)
{
    long long i = (long long)blockIdx.x * 256 + threadIdx.x;
    if (i < n) p[i] = v;
}

// ---------------------------------------------------------------------------
// Host orchestration
// ---------------------------------------------------------------------------
static inline void launch_gemm(const float* X, long long ldx, const float* W,
                               const float* bias, float* Y1, long long ldy1,
                               float* Y2, long long ldy2,
                               const float* resid, long long ldr,
                               const float* skipv, int nrows,
                               int act_in, int act_out, hipStream_t s)
{
    dim3 grid((unsigned)((nrows + 31) / 32));
    gemm128_kernel<<<grid, 256, 0, s>>>(X, ldx, W, bias, Y1, ldy1, Y2, ldy2,
                                        resid, ldr, skipv, nrows, act_in, act_out);
}

static inline void launch_fill(void* p, unsigned int v, long long n, hipStream_t s)
{
    dim3 grid((unsigned)((n + 255) / 256));
    fill_kernel<<<grid, 256, 0, s>>>((unsigned int*)p, v, n);
}

extern "C" void kernel_launch(void* const* d_in, const int* in_sizes, int n_in,
                              void* d_out, int out_size, void* d_ws, size_t ws_size,
                              hipStream_t stream)
{
    (void)in_sizes; (void)n_in; (void)out_size; (void)ws_size;

    const float* x0   = (const float*)d_in[0];
    const float* x1   = (const float*)d_in[1];
    const int* esrc[3] = {(const int*)d_in[2], (const int*)d_in[4], (const int*)d_in[6]};
    const int* edst[3] = {(const int*)d_in[3], (const int*)d_in[5], (const int*)d_in[7]};
    const float* W_in = (const float*)d_in[8];
    const float* b_in = (const float*)d_in[9];
    const float* Wk   = (const float*)d_in[10];
    const float* bk   = (const float*)d_in[11];
    const float* Wq   = (const float*)d_in[12];
    const float* bq   = (const float*)d_in[13];
    const float* Wv   = (const float*)d_in[14];
    const float* bv   = (const float*)d_in[15];
    const float* Wa   = (const float*)d_in[16];
    const float* ba   = (const float*)d_in[17];
    const float* skip = (const float*)d_in[18];
    const float* a_rel = (const float*)d_in[19];
    const float* m_rel = (const float*)d_in[20];
    const float* p_rel = (const float*)d_in[21];

    const int N0 = 100000, N1 = 50000, E = 200000;
    const int NT[2] = {N0, N1};
    const int ST[3] = {0, 1, 1};
    const int DT[3] = {1, 0, 1};

    float* y0 = (float*)d_out;                       // [N0][384]
    float* y1 = y0 + (size_t)N0 * 384;               // [N1][384]

    // workspace carve (floats)
    float* w = (float*)d_ws;
    float* h[2];   h[0] = w;  w += (size_t)N0 * 128;
                   h[1] = w;  w += (size_t)N1 * 128;
    float* q[2];   q[0] = w;  w += (size_t)N0 * 128;
                   q[1] = w;  w += (size_t)N1 * 128;
    float* kr  = w;  w += (size_t)N0 * 128;
    float* vr  = w;  w += (size_t)N0 * 128;
    float* agg[2]; agg[0] = w; w += (size_t)N0 * 128;
                   agg[1] = w; w += (size_t)N1 * 128;
    float* alpha = w; w += (size_t)E * 4;
    float* asum  = w; w += (size_t)N0 * 4;
    unsigned int* amaxKey = (unsigned int*)w; w += (size_t)N0 * 4;
    float* Wck = w; w += 16384;  float* bck = w; w += 128;
    float* Wcv = w; w += 16384;  float* bcv = w; w += 128;

    // ---- input projection: h_t = relu(x_t @ W_in[t] + b_in[t]); also y[:,0:128]
    launch_gemm(x0, 128, W_in,          b_in,        h[0], 128, y0, 384,
                nullptr, 0, nullptr, N0, 0, 1, stream);
    launch_gemm(x1, 128, W_in + 16384,  b_in + 128,  h[1], 128, y1, 384,
                nullptr, 0, nullptr, N1, 0, 1, stream);

    for (int l = 0; l < 2; ++l) {
        // q projections
        for (int t = 0; t < 2; ++t)
            launch_gemm(h[t], 128, Wq + (size_t)(l * 2 + t) * 16384,
                        bq + (size_t)(l * 2 + t) * 128, q[t], 128,
                        nullptr, 0, nullptr, 0, nullptr, NT[t], 0, 0, stream);

        launch_fill(agg[0], 0u, (long long)N0 * 128, stream);
        launch_fill(agg[1], 0u, (long long)N1 * 128, stream);

        for (int r = 0; r < 3; ++r) {
            const int st = ST[r], dt = DT[r];
            const float* ar = a_rel + (size_t)(l * 3 + r) * 4096;
            const float* mr = m_rel + (size_t)(l * 3 + r) * 4096;

            compose_kernel<<<65, 256, 0, stream>>>(
                Wk + (size_t)(l * 2 + st) * 16384, bk + (size_t)(l * 2 + st) * 128,
                ar, Wck, bck);
            compose_kernel<<<65, 256, 0, stream>>>(
                Wv + (size_t)(l * 2 + st) * 16384, bv + (size_t)(l * 2 + st) * 128,
                mr, Wcv, bcv);

            // kr = h[st] @ Wck + bck ; vr = h[st] @ Wcv + bcv   (WMMA GEMMs)
            launch_gemm(h[st], 128, Wck, bck, kr, 128, nullptr, 0,
                        nullptr, 0, nullptr, NT[st], 0, 0, stream);
            launch_gemm(h[st], 128, Wcv, bcv, vr, 128, nullptr, 0,
                        nullptr, 0, nullptr, NT[st], 0, 0, stream);

            const long long segn = (long long)NT[dt] * 4;
            launch_fill(amaxKey, 0u, segn, stream);
            launch_fill(asum,    0u, segn, stream);

            const int nEH = E * 4;
            edge_alpha_kernel<<<(nEH + 255) / 256, 256, 0, stream>>>(
                esrc[r], edst[r], q[dt], kr, p_rel + (size_t)(l * 3 + r) * 4,
                alpha, amaxKey, E);
            edge_exp_kernel<<<(nEH + 255) / 256, 256, 0, stream>>>(
                edst[r], alpha, amaxKey, asum, E);
            const long long nEC = (long long)E * 128;
            edge_scatter_kernel<<<(unsigned)((nEC + 255) / 256), 256, 0, stream>>>(
                esrc[r], edst[r], vr, alpha, asum, agg[dt], E);
        }

        // out: hnew = a*(gelu(agg)@Wa + ba) + (1-a)*h ; also y[:, (l+1)*128 ..]
        for (int t = 0; t < 2; ++t) {
            float* ydst = (t == 0 ? y0 : y1) + (size_t)(l + 1) * 128;
            launch_gemm(agg[t], 128, Wa + (size_t)(l * 2 + t) * 16384,
                        ba + (size_t)(l * 2 + t) * 128,
                        h[t], 128, ydst, 384,
                        h[t], 128, skip + (l * 2 + t), NT[t], 2, 0, stream);
        }
    }
}
